// GAT_5514738008403
// MI455X (gfx1250) — matrix operand order repaired and checked by reading, not compiled, not run on hardware
//
#include <hip/hip_runtime.h>
#include <hip/hip_bf16.h>
#include <stdint.h>

// ---------------------------------------------------------------------------
// GAT (2-layer, 8 heads) for MI455X / gfx1250.
// GEMMs run on v_wmma_f32_16x16x32_bf16; each wave computes a 16x64 C strip
// (4 accumulators, A fragment reused 4x). All fragment loads for a K-step are
// batched before the WMMAs so the backend can overlap loads with matrix ops
// via partial s_wait_loadcnt.
// Edge segment-softmax passes use ordered-uint atomicMax + f32 atomicAdd.
// ---------------------------------------------------------------------------

typedef __attribute__((ext_vector_type(16))) __bf16 v16bf;
typedef __attribute__((ext_vector_type(8)))  float  v8f;

#define HEADS 8
#define HD1   32
#define NCLS  40
#define FIN   128
#define M1    (HEADS * HD1)   // 256
#define M2    (HEADS * NCLS)  // 320
#define NEG_SLOPE 0.2f
#define NBLK  4               // feature tiles per wave (16 and 20 both % 4 == 0)

// ---- helpers ---------------------------------------------------------------

__device__ __forceinline__ unsigned short f32_to_bf16(float f) {
  unsigned u = __float_as_uint(f);
  u += 0x7FFFu + ((u >> 16) & 1u);          // round to nearest even
  return (unsigned short)(u >> 16);
}

// monotone float <-> uint mapping so atomicMax(uint) == float max
__device__ __forceinline__ unsigned f32_ord(float f) {
  unsigned u = __float_as_uint(f);
  return (u & 0x80000000u) ? ~u : (u | 0x80000000u);
}
__device__ __forceinline__ float ord_f32(unsigned o) {
  return (o & 0x80000000u) ? __uint_as_float(o & 0x7FFFFFFFu)
                           : __uint_as_float(~o);
}

// ---- small utility kernels -------------------------------------------------

__global__ void cvt_bf16_kernel(const float* __restrict__ s,
                                unsigned short* __restrict__ d, int n) {
  int i = blockIdx.x * blockDim.x + threadIdx.x;
  if (i < n) d[i] = f32_to_bf16(s[i]);
}

__global__ void zero_u32_kernel(unsigned* __restrict__ p, int n) {
  int i = blockIdx.x * blockDim.x + threadIdx.x;
  if (i < n) p[i] = 0u;
}

// ---- WMMA GEMM:  C[N,M] = A[N,K] * B[M,K]^T   (A,B bf16 row-major) ---------
// One wave per 16x(16*NBLK) C strip; K stepped by 32; A fragment reused NBLK x.

union FragBF { v16bf v; uint4 q[2]; };

__global__ __launch_bounds__(256) void gemm_bf16_wmma(
    const unsigned short* __restrict__ A,   // [N,K] bf16
    const unsigned short* __restrict__ B,   // [M,K] bf16
    float* __restrict__ C,                  // [N,M] f32
    int N, int M, int K) {
  int wave   = blockIdx.x * (blockDim.x >> 5) + (threadIdx.x >> 5);
  int tilesM = M >> 4;
  int groups = tilesM / NBLK;
  int total  = (N >> 4) * groups;
  if (wave >= total) return;                // wave-uniform: EXEC all-1s below
  int tn  = wave / groups;                  // node-row tile
  int tg  = wave % groups;                  // feature-tile group (NBLK tiles)
  int lane = threadIdx.x & 31;
  int half = lane >> 4;                     // 0: lanes 0-15, 1: lanes 16-31
  int lr   = lane & 15;

  const unsigned short* arow = A + (size_t)(tn * 16 + lr) * K;
  const unsigned short* brow[NBLK];
#pragma unroll
  for (int j = 0; j < NBLK; ++j)
    brow[j] = B + (size_t)((tg * NBLK + j) * 16 + lr) * K;

  v8f acc[NBLK];
#pragma unroll
  for (int j = 0; j < NBLK; ++j) acc[j] = (v8f){};

  for (int k0 = 0; k0 < K; k0 += 32) {
    // Batch ALL fragment loads first so the backend can clause them and
    // overlap with the WMMAs below using partial s_wait_loadcnt.
    FragBF a, b[NBLK];
    // A 16x32: lanes<16 -> K {0..7,16..23}; lanes>=16 -> K {8..15,24..31}
    a.q[0] = *(const uint4*)(arow + k0 + half * 8);
    a.q[1] = *(const uint4*)(arow + k0 + 16 + half * 8);
#pragma unroll
    for (int j = 0; j < NBLK; ++j) {
      // B 32x16: lanes<16 -> K {0..7,16..23} of column lr; lanes>=16 -> K {8..15,24..31}
      b[j].q[0] = *(const uint4*)(brow[j] + k0 + half * 8);
      b[j].q[1] = *(const uint4*)(brow[j] + k0 + 16 + half * 8);
    }
#pragma unroll
    for (int j = 0; j < NBLK; ++j)
      acc[j] = __builtin_amdgcn_wmma_f32_16x16x32_bf16(
          false, a.v, false, b[j].v, (short)0, acc[j], false, false);
  }
  // C/D layout: VGPR i -> row (half*8 + i), col = lr
  int rbase = tn * 16 + half * 8;
#pragma unroll
  for (int j = 0; j < NBLK; ++j) {
    int cc = (tg * NBLK + j) * 16 + lr;
#pragma unroll
    for (int i = 0; i < 8; ++i)
      C[(size_t)(rbase + i) * M + cc] = acc[j][i];
  }
}

// ---- per-node attention logits:  al[n,h] = dot(h[n,h,:], a[h,:]) -----------

__global__ void attn_coef_kernel(const float* __restrict__ h,
                                 const float* __restrict__ a_src,
                                 const float* __restrict__ a_dst,
                                 float* __restrict__ als,
                                 float* __restrict__ ald, int N, int C) {
  int i = blockIdx.x * blockDim.x + threadIdx.x;
  if (i >= N * HEADS) return;
  int n = i >> 3, hh = i & 7;
  const float* hp = h + (size_t)n * HEADS * C + hh * C;
  const float* as = a_src + hh * C;
  const float* ad = a_dst + hh * C;
  float ss = 0.f, sd = 0.f;
  for (int c = 0; c < C; ++c) {
    float v = hp[c];
    ss += v * as[c];
    sd += v * ad[c];
  }
  als[i] = ss;
  ald[i] = sd;
}

// ---- edge pass 1: segment max of leaky_relu(al_src[s]+al_dst[d]) -----------

__global__ void edge_max_kernel(const long long* __restrict__ ei, int E, int N,
                                const float* __restrict__ als,
                                const float* __restrict__ ald,
                                unsigned* __restrict__ m) {
  int i  = blockIdx.x * blockDim.x + threadIdx.x;
  int ET = E + N;
  if (i >= ET * HEADS) return;
  int e = i >> 3, hh = i & 7;
  int s, d;
  if (e < E) { s = (int)ei[e]; d = (int)ei[E + e]; }
  else       { s = d = e - E; }               // self-loops
  float v = als[s * HEADS + hh] + ald[d * HEADS + hh];
  v = v > 0.f ? v : NEG_SLOPE * v;
  atomicMax(&m[d * HEADS + hh], f32_ord(v));
}

// ---- edge pass 2: denom += exp(e-m);  accum[dst] += exp(e-m)*h[src] --------

__global__ void edge_acc_kernel(const long long* __restrict__ ei, int E, int N,
                                const float* __restrict__ als,
                                const float* __restrict__ ald,
                                const unsigned* __restrict__ m,
                                const float* __restrict__ h,
                                float* __restrict__ denom,
                                float* __restrict__ accum, int C) {
  int i  = blockIdx.x * blockDim.x + threadIdx.x;
  int ET = E + N;
  if (i >= ET * HEADS) return;
  int e = i >> 3, hh = i & 7;
  int s, d;
  if (e < E) { s = (int)ei[e]; d = (int)ei[E + e]; }
  else       { s = d = e - E; }
  float v = als[s * HEADS + hh] + ald[d * HEADS + hh];
  v = v > 0.f ? v : NEG_SLOPE * v;
  float mv = ord_f32(m[d * HEADS + hh]);
  float ee = __expf(v - mv);
  atomicAdd(&denom[d * HEADS + hh], ee);
  const float* hs = h + (size_t)s * HEADS * C + hh * C;
  float* ac       = accum + (size_t)d * HEADS * C + hh * C;
  for (int c = 0; c < C; ++c) atomicAdd(&ac[c], ee * hs[c]);
}

// ---- layer-1 finalize: divide, +bias, ELU, emit bf16 for GEMM2 -------------

__global__ void finalize1_kernel(const float* __restrict__ accum,
                                 const float* __restrict__ denom,
                                 const float* __restrict__ b1,
                                 unsigned short* __restrict__ hb, int N) {
  int i = blockIdx.x * blockDim.x + threadIdx.x;
  if (i >= N * M1) return;
  int n = i / M1, f = i % M1, hh = f / HD1;
  float v = accum[i] / denom[n * HEADS + hh] + b1[f];
  v = v > 0.f ? v : (__expf(v) - 1.f);        // ELU
  hb[i] = f32_to_bf16(v);
}

// ---- layer-2 finalize: divide, mean over heads, +bias, log_softmax ---------

__global__ void finalize2_kernel(const float* __restrict__ accum,
                                 const float* __restrict__ denom,
                                 const float* __restrict__ b2,
                                 float* __restrict__ out, int N) {
  int n = blockIdx.x * blockDim.x + threadIdx.x;
  if (n >= N) return;
  float o[NCLS];
#pragma unroll
  for (int c = 0; c < NCLS; ++c) o[c] = 0.f;
  for (int hh = 0; hh < HEADS; ++hh) {
    float inv       = 1.f / denom[n * HEADS + hh];
    const float* ap = accum + (size_t)n * M2 + hh * NCLS;
#pragma unroll
    for (int c = 0; c < NCLS; ++c) o[c] += ap[c] * inv;
  }
  float mx = -1e30f;
#pragma unroll
  for (int c = 0; c < NCLS; ++c) {
    o[c] = o[c] * (1.f / HEADS) + b2[c];
    mx   = fmaxf(mx, o[c]);
  }
  float s = 0.f;
#pragma unroll
  for (int c = 0; c < NCLS; ++c) s += __expf(o[c] - mx);
  float l = mx + __logf(s);
#pragma unroll
  for (int c = 0; c < NCLS; ++c) out[(size_t)n * NCLS + c] = o[c] - l;
}

// ---------------------------------------------------------------------------

extern "C" void kernel_launch(void* const* d_in, const int* in_sizes, int n_in,
                              void* d_out, int out_size, void* d_ws,
                              size_t ws_size, hipStream_t stream) {
  (void)n_in; (void)out_size; (void)ws_size;
  const float*     x   = (const float*)d_in[0];
  const long long* ei  = (const long long*)d_in[1];   // int64 [2,E]
  const float*     W1  = (const float*)d_in[2];
  const float*     as1 = (const float*)d_in[3];
  const float*     ad1 = (const float*)d_in[4];
  const float*     b1  = (const float*)d_in[5];
  const float*     W2  = (const float*)d_in[6];
  const float*     as2 = (const float*)d_in[7];
  const float*     ad2 = (const float*)d_in[8];
  const float*     b2  = (const float*)d_in[9];
  float* out = (float*)d_out;

  int N  = in_sizes[0] / FIN;   // 50000 (multiple of 16)
  int E  = in_sizes[1] / 2;     // 800000
  int ET = E + N;

  // ---- carve workspace -----------------------------------------------------
  char* p = (char*)d_ws;
  auto carve = [&](size_t bytes) {
    char* r = p;
    p += (bytes + 255) & ~(size_t)255;
    return (void*)r;
  };
  unsigned short* xb   = (unsigned short*)carve((size_t)N * FIN * 2);
  unsigned short* w1b  = (unsigned short*)carve((size_t)M1 * FIN * 2);
  unsigned short* w2b  = (unsigned short*)carve((size_t)M2 * M1 * 2);
  float*          h1   = (float*)carve((size_t)N * M1 * 4);
  float*          als_ = (float*)carve((size_t)N * HEADS * 4);
  float*          ald_ = (float*)carve((size_t)N * HEADS * 4);
  unsigned*       mbuf = (unsigned*)carve((size_t)N * HEADS * 4);
  float*          den  = (float*)carve((size_t)N * HEADS * 4);
  float*          acc1 = (float*)carve((size_t)N * M1 * 4);
  unsigned short* hb   = (unsigned short*)carve((size_t)N * M1 * 2);
  float*          h2   = (float*)carve((size_t)N * M2 * 4);
  float*          acc2 = (float*)carve((size_t)N * M2 * 4);

  const int BT = 256;
  auto blocks = [](int n, int bt) { return (unsigned)((n + bt - 1) / bt); };

  // ---- bf16 conversions ----------------------------------------------------
  cvt_bf16_kernel<<<blocks(N * FIN, BT), BT, 0, stream>>>(x, xb, N * FIN);
  cvt_bf16_kernel<<<blocks(M1 * FIN, BT), BT, 0, stream>>>(W1, w1b, M1 * FIN);
  cvt_bf16_kernel<<<blocks(M2 * M1, BT), BT, 0, stream>>>(W2, w2b, M2 * M1);

  // ---- layer 1 -------------------------------------------------------------
  int waves1 = (N / 16) * (M1 / 16 / NBLK);
  gemm_bf16_wmma<<<(unsigned)((waves1 + 7) / 8), 256, 0, stream>>>(
      xb, w1b, h1, N, M1, FIN);
  attn_coef_kernel<<<blocks(N * HEADS, BT), BT, 0, stream>>>(
      h1, as1, ad1, als_, ald_, N, HD1);
  zero_u32_kernel<<<blocks(N * HEADS, BT), BT, 0, stream>>>(mbuf, N * HEADS);
  zero_u32_kernel<<<blocks(N * HEADS, BT), BT, 0, stream>>>((unsigned*)den, N * HEADS);
  zero_u32_kernel<<<blocks(N * M1, BT), BT, 0, stream>>>((unsigned*)acc1, N * M1);
  edge_max_kernel<<<blocks(ET * HEADS, BT), BT, 0, stream>>>(
      ei, E, N, als_, ald_, mbuf);
  edge_acc_kernel<<<blocks(ET * HEADS, BT), BT, 0, stream>>>(
      ei, E, N, als_, ald_, mbuf, h1, den, acc1, HD1);
  finalize1_kernel<<<blocks(N * M1, BT), BT, 0, stream>>>(acc1, den, b1, hb, N);

  // ---- layer 2 -------------------------------------------------------------
  int waves2 = (N / 16) * (M2 / 16 / NBLK);
  gemm_bf16_wmma<<<(unsigned)((waves2 + 7) / 8), 256, 0, stream>>>(
      hb, w2b, h2, N, M2, M1);
  attn_coef_kernel<<<blocks(N * HEADS, BT), BT, 0, stream>>>(
      h2, as2, ad2, als_, ald_, N, NCLS);
  zero_u32_kernel<<<blocks(N * HEADS, BT), BT, 0, stream>>>(mbuf, N * HEADS);
  zero_u32_kernel<<<blocks(N * HEADS, BT), BT, 0, stream>>>((unsigned*)den, N * HEADS);
  zero_u32_kernel<<<blocks(N * M2, BT), BT, 0, stream>>>((unsigned*)acc2, N * M2);
  edge_max_kernel<<<blocks(ET * HEADS, BT), BT, 0, stream>>>(
      ei, E, N, als_, ald_, mbuf);
  edge_acc_kernel<<<blocks(ET * HEADS, BT), BT, 0, stream>>>(
      ei, E, N, als_, ald_, mbuf, h2, den, acc2, NCLS);
  finalize2_kernel<<<blocks(N, BT), BT, 0, stream>>>(acc2, den, b2, out, N);
}
